// DGCNNEncoder_24343874634076
// MI455X (gfx1250) — matrix operand order repaired and checked by reading, not compiled, not run on hardware
//
#include <hip/hip_runtime.h>

typedef __attribute__((ext_vector_type(16))) _Float16 v16h;
typedef __attribute__((ext_vector_type(8)))  _Float16 v8h;
typedef __attribute__((ext_vector_type(8)))  float    v8f;

#define NPT   2048
#define KNN   20
#define NROWS (NPT * KNN)   // 40960 flattened (n,k) rows
#define HTLD  448           // hT row stride: 64 + 128 + 256
#define EPS   1e-5f

// ---------------- WMMA fragment loaders (ISA 7.12.2 layouts) ----------------
// A (16x32 f16, MxK): lanes 0-15 row M=lane hold K={k0..k0+7, k0+16..k0+23};
// lanes 16-31 row M=lane-16 hold K={k0+8..k0+15, k0+24..k0+31}.
__device__ __forceinline__ v16h frag_a(const _Float16* p, int ld, int row0, int k0, int lane) {
    const _Float16* q = p + (size_t)(row0 + (lane & 15)) * ld + k0 + ((lane >> 4) << 3);
    v16h a;
#pragma unroll
    for (int i = 0; i < 8; ++i) a[i] = q[i];
#pragma unroll
    for (int i = 0; i < 8; ++i) a[8 + i] = q[16 + i];
    return a;
}

// B (32x16 f16, KxN), from a transposed buffer bt where B[k][n] = bt[n*ld + k]:
// lanes 0-15 col N=lane hold K={k0..k0+7, k0+16..k0+23}; lanes 16-31 col N=lane-16 hold K={k0+8..k0+15, k0+24..k0+31}.
__device__ __forceinline__ v16h frag_b(const _Float16* bt, int ld, int k0, int n0, int lane) {
    const _Float16* q = bt + (size_t)(n0 + (lane & 15)) * ld + k0 + ((lane >> 4) << 3);
    v16h b;
#pragma unroll
    for (int i = 0; i < 8; ++i) b[i] = q[i];
#pragma unroll
    for (int i = 0; i < 8; ++i) b[8 + i] = q[16 + i];
    return b;
}

// C/D f32 16x16: VGPR r -> lanes 0-15: (M=r, N=lane); lanes 16-31: (M=r+8, N=lane-16).

// ---------------- small utility kernels ----------------
__global__ void k_zero(float* p, int n) {
    int i = blockIdx.x * 256 + threadIdx.x;
    if (i < n) p[i] = 0.0f;
}

__global__ void k_cvt_w(const float* __restrict__ w, _Float16* __restrict__ wh,
                        int O, int Cin, int Cpad) {
    int i = blockIdx.x * 256 + threadIdx.x;
    if (i >= O * Cpad) return;
    int o = i / Cpad, c = i % Cpad;
    wh[i] = (c < Cin) ? (_Float16)w[o * Cin + c] : (_Float16)0.0f;
}

__global__ void k_pack_x0(const float* __restrict__ x, _Float16* __restrict__ xt0,
                          float* __restrict__ xx) {
    int n = blockIdx.x * 256 + threadIdx.x;
    if (n >= NPT) return;
    float a = x[n], b = x[NPT + n], c = x[2 * NPT + n];
    _Float16* q = xt0 + n * 32;
    q[0] = (_Float16)a; q[1] = (_Float16)b; q[2] = (_Float16)c;
#pragma unroll
    for (int j = 3; j < 32; ++j) q[j] = (_Float16)0.0f;
    xx[n] = a * a + b * b + c * c;
}

__global__ void k_sumsq(const _Float16* __restrict__ ht, int C, float* __restrict__ xx) {
    int n = blockIdx.x * 256 + threadIdx.x;
    if (n >= NPT) return;
    const _Float16* p = ht + (size_t)n * HTLD;
    float s = 0.0f;
    for (int c = 0; c < C; ++c) { float v = (float)p[c]; s += v * v; }
    xx[n] = s;
}

// ---------------- distance Gram matrix via WMMA, LDS-staged B panel ----------------
// dist[n][m] = 2 * <x_n, x_m> - xx[n] - xx[m]
// Block: 8 waves; waves cover 8 m-tiles; B panel (128 cols x C) staged once in LDS.
template <int C>
__global__ __launch_bounds__(256) void k_dist_gemm(const _Float16* __restrict__ xt, int ld,
                                                   const float* __restrict__ xx,
                                                   float* __restrict__ dist) {
    constexpr int LDB = C + 8;  // +8 halves: break power-of-two bank stride
    __shared__ _Float16 Bsh[128 * LDB];
    const int tid = threadIdx.x, lane = tid & 31, wave = tid >> 5;
    const int nc = blockIdx.y * 128;
    for (int i = tid; i < 128 * (C / 8); i += 256) {
        int row = i / (C / 8), cp = (i % (C / 8)) * 8;
        *(v8h*)(&Bsh[row * LDB + cp]) = *(const v8h*)(xt + (size_t)(nc + row) * ld + cp);
    }
    __syncthreads();

    const int m0 = (blockIdx.x * 8 + wave) * 16;
    constexpr int S = C / 32;
    v16h A[S];
#pragma unroll
    for (int s = 0; s < S; ++s) A[s] = frag_a(xt, ld, m0, s * 32, lane);
    const int col = lane & 15, rb = (lane >> 4) << 3;
    for (int t = 0; t < 8; ++t) {
        int n0l = t * 16;
        v8f acc = {};
#pragma unroll
        for (int s = 0; s < S; ++s)
            acc = __builtin_amdgcn_wmma_f32_16x16x32_f16(
                false, A[s], false, frag_b(Bsh, LDB, s * 32, n0l, lane),
                (short)0, acc, false, false);
        float xxn = xx[nc + n0l + col];
#pragma unroll
        for (int r = 0; r < 8; ++r) {
            int m = m0 + rb + r;
            dist[(size_t)m * NPT + nc + n0l + col] = 2.0f * acc[r] - xx[m] - xxn;
        }
    }
}

// ---------------- top-K selection (one wave per point) ----------------
__device__ __forceinline__ void topk_rounds(float* row, int* out, int lane) {
    for (int r = 0; r < KNN; ++r) {
        float best = -3.0e38f; int bi = 0x7fffffff;
        for (int m = lane; m < NPT; m += 32) {
            float v = row[m];
            if (v > best) { best = v; bi = m; }
        }
#pragma unroll
        for (int off = 16; off > 0; off >>= 1) {
            float ob = __shfl_xor(best, off, 32);
            int   oi = __shfl_xor(bi, off, 32);
            if (ob > best || (ob == best && oi < bi)) { best = ob; bi = oi; }
        }
        if (lane == 0) { out[r] = bi; row[bi] = -3.0e38f; }
        __syncthreads();
    }
}

__global__ __launch_bounds__(32) void k_topk_l1(const float* __restrict__ x,
                                                const float* __restrict__ xx, int* idx) {
    __shared__ float row[NPT];
    int n = blockIdx.x, lane = threadIdx.x;
    float a = x[n], b = x[NPT + n], c = x[2 * NPT + n], xxn = xx[n];
    for (int m = lane; m < NPT; m += 32)
        row[m] = 2.0f * (a * x[m] + b * x[NPT + m] + c * x[2 * NPT + m]) - xxn - xx[m];
    __syncthreads();
    topk_rounds(row, idx + n * KNN, lane);
}

__global__ __launch_bounds__(32) void k_topk(const float* __restrict__ dist, int* idx) {
    __shared__ float row[NPT];
    int n = blockIdx.x, lane = threadIdx.x;
    const float* d = dist + (size_t)n * NPT;
    for (int m = lane; m < NPT; m += 32) row[m] = d[m];
    __syncthreads();
    topk_rounds(row, idx + n * KNN, lane);
}

// ---------------- gathered edge features (vectorized): F[r=n*K+k][c] ----------------
template <int C, int CF>
__global__ void k_feat(const _Float16* __restrict__ xt, int ld,
                       const int* __restrict__ idx, _Float16* __restrict__ F) {
    constexpr int CH = CF / 8;  // 16B chunks per row
    int gid = blockIdx.x * 256 + threadIdx.x;
    if (gid >= NROWS * CH) return;
    int r = gid / CH, c = (gid % CH) * 8;
    int n = r / KNN, k = r % KNN;
    int j = idx[n * KNN + k];
    v8h v;
    if (c + 8 <= C) {                       // fully inside (nbr - ctr)
        v8h a = *(const v8h*)(xt + (size_t)j * ld + c);
        v8h b = *(const v8h*)(xt + (size_t)n * ld + c);
        v = a - b;
    } else if (c >= C && c + 8 <= 2 * C) {  // fully inside ctr
        v = *(const v8h*)(xt + (size_t)n * ld + (c - C));
    } else {                                // boundary / pad (layer-1 only)
#pragma unroll
        for (int e = 0; e < 8; ++e) {
            int cc = c + e;
            _Float16 t;
            if (cc < C)          t = xt[(size_t)j * ld + cc] - xt[(size_t)n * ld + cc];
            else if (cc < 2 * C) t = xt[(size_t)n * ld + (cc - C)];
            else                 t = (_Float16)0.0f;
            v[e] = t;
        }
    }
    *(v8h*)(F + (size_t)r * CF + c) = v;
}

// ---------------- edge-conv GEMM with LDS-staged weights ----------------
// y[r][o] = sum_c F[r][c] * w[o][c].  Block: 8 waves = 8 m-tiles; weights staged
// in 64-output-row groups (<= 34 KB LDS) and shared by all waves.
template <int KF, int O>
__global__ __launch_bounds__(256) void k_edge_gemm(const _Float16* __restrict__ F,
                                                   const _Float16* __restrict__ wh,
                                                   float* __restrict__ y) {
    constexpr int LDW = KF + 8;
    __shared__ _Float16 wsh[64 * LDW];
    const int tid = threadIdx.x, lane = tid & 31, wave = tid >> 5;
    const int m0 = (blockIdx.x * 8 + wave) * 16;
    constexpr int S = KF / 32;
    v16h A[S];
#pragma unroll
    for (int s = 0; s < S; ++s) A[s] = frag_a(F, KF, m0, s * 32, lane);
    const int col = lane & 15, rb = (lane >> 4) << 3;

    for (int og = 0; og < O / 64; ++og) {
        for (int i = tid; i < 64 * (KF / 8); i += 256) {
            int row = i / (KF / 8), cp = (i % (KF / 8)) * 8;
            *(v8h*)(&wsh[row * LDW + cp]) =
                *(const v8h*)(wh + (size_t)(og * 64 + row) * KF + cp);
        }
        __syncthreads();
        for (int tl = 0; tl < 4; ++tl) {
            v8f acc = {};
#pragma unroll
            for (int s = 0; s < S; ++s)
                acc = __builtin_amdgcn_wmma_f32_16x16x32_f16(
                    false, A[s], false, frag_b(wsh, LDW, s * 32, tl * 16, lane),
                    (short)0, acc, false, false);
            int o0 = og * 64 + tl * 16;
#pragma unroll
            for (int r = 0; r < 8; ++r)
                y[(size_t)(m0 + rb + r) * O + o0 + col] = acc[r];
        }
        __syncthreads();
    }
}

// ---------------- BN + ReLU + max over K, write f16 hT slice ----------------
__global__ void k_bn_max(const float* __restrict__ y, int O, const float* __restrict__ bb,
                         const float* __restrict__ g, const float* __restrict__ be,
                         const float* __restrict__ mn, const float* __restrict__ vr,
                         _Float16* __restrict__ htdst) {
    int gid = blockIdx.x * 256 + threadIdx.x;
    if (gid >= NPT * O) return;
    int n = gid / O, o = gid % O;
    float sc = g[o] * rsqrtf(vr[o] + EPS);
    float sh = (bb[o] - mn[o]) * sc + be[o];
    float mx = 0.0f;  // ReLU folds into max with 0 seed
    for (int k = 0; k < KNN; ++k)
        mx = fmaxf(mx, y[(size_t)(n * KNN + k) * O + o] * sc + sh);
    htdst[(size_t)n * HTLD + o] = (_Float16)mx;
}

// ---------------- final projection + BN + ReLU + max over N ----------------
// Block: 8 waves, each owning one o-tile with resident wo A-fragments.
// hT is staged cooperatively into LDS in 64-column panels (58 KB), so the
// WMMA chain depends only on short LDS loads, never on global latency.
__global__ __launch_bounds__(256) void k_final(const _Float16* __restrict__ woh,
                                               const _Float16* __restrict__ ht,
                                               const float* __restrict__ bo,
                                               const float* __restrict__ go,
                                               const float* __restrict__ beo,
                                               const float* __restrict__ mo,
                                               const float* __restrict__ vo,
                                               float* __restrict__ out) {
    constexpr int LDH = HTLD + 8;  // 456 halves = 228 dwords: no 64-bank stride
    __shared__ _Float16 Hsh[64 * LDH];
    const int tid = threadIdx.x, lane = tid & 31, wave = tid >> 5;
    const int o0 = (blockIdx.x * 8 + wave) * 16;
    v16h A[14];  // 448/32 k-steps of wo rows, held resident
#pragma unroll
    for (int s = 0; s < 14; ++s) A[s] = frag_a(woh, HTLD, o0, s * 32, lane);
    const int rb = (lane >> 4) << 3;
    float sc[8], sh[8], mx[8];
#pragma unroll
    for (int r = 0; r < 8; ++r) {
        int o = o0 + rb + r;
        sc[r] = go[o] * rsqrtf(vo[o] + EPS);
        sh[r] = (bo[o] - mo[o]) * sc[r] + beo[o];
        mx[r] = 0.0f;
    }
    const int nbase = blockIdx.y * 256;
    for (int nb = 0; nb < 4; ++nb) {            // 64-column sub-panels
        for (int i = tid; i < 64 * (HTLD / 8); i += 256) {
            int row = i / (HTLD / 8), cp = (i % (HTLD / 8)) * 8;
            *(v8h*)(&Hsh[row * LDH + cp]) =
                *(const v8h*)(ht + (size_t)(nbase + nb * 64 + row) * HTLD + cp);
        }
        __syncthreads();
        for (int tl = 0; tl < 4; ++tl) {        // 4 n-tiles per panel
            v8f acc = {};
#pragma unroll
            for (int s = 0; s < 14; ++s)
                acc = __builtin_amdgcn_wmma_f32_16x16x32_f16(
                    false, A[s], false, frag_b(Hsh, LDH, s * 32, tl * 16, lane),
                    (short)0, acc, false, false);
#pragma unroll
            for (int r = 0; r < 8; ++r)
                mx[r] = fmaxf(mx[r], acc[r] * sc[r] + sh[r]);
        }
        __syncthreads();
    }
#pragma unroll
    for (int r = 0; r < 8; ++r) {
#pragma unroll
        for (int off = 8; off > 0; off >>= 1)
            mx[r] = fmaxf(mx[r], __shfl_xor(mx[r], off, 32));
    }
    if ((lane & 15) == 0) {
        // post-ReLU values >= 0 -> IEEE bits monotonic; int atomicMax is exact.
#pragma unroll
        for (int r = 0; r < 8; ++r)
            atomicMax((int*)&out[o0 + rb + r], __float_as_int(mx[r]));
    }
}

// ---------------- host orchestration ----------------
extern "C" void kernel_launch(void* const* d_in, const int* in_sizes, int n_in,
                              void* d_out, int out_size, void* d_ws, size_t ws_size,
                              hipStream_t stream) {
    (void)in_sizes; (void)n_in; (void)ws_size;
    const float* x   = (const float*)d_in[0];
    const float* w1  = (const float*)d_in[1];
    const float* b1  = (const float*)d_in[2];
    const float* g1  = (const float*)d_in[3];
    const float* be1 = (const float*)d_in[4];
    const float* m1  = (const float*)d_in[5];
    const float* v1  = (const float*)d_in[6];
    const float* w2  = (const float*)d_in[7];
    const float* b2  = (const float*)d_in[8];
    const float* g2  = (const float*)d_in[9];
    const float* be2 = (const float*)d_in[10];
    const float* m2  = (const float*)d_in[11];
    const float* v2  = (const float*)d_in[12];
    const float* w3  = (const float*)d_in[13];
    const float* b3  = (const float*)d_in[14];
    const float* g3  = (const float*)d_in[15];
    const float* be3 = (const float*)d_in[16];
    const float* m3  = (const float*)d_in[17];
    const float* v3  = (const float*)d_in[18];
    const float* wo  = (const float*)d_in[19];
    const float* bo  = (const float*)d_in[20];
    const float* go  = (const float*)d_in[21];
    const float* beo = (const float*)d_in[22];
    const float* mo  = (const float*)d_in[23];
    const float* vo  = (const float*)d_in[24];

    char* wp = (char*)d_ws;
    auto carve = [&](size_t bytes) -> char* {
        char* p = wp;
        wp += (bytes + 255) & ~(size_t)255;
        return p;
    };
    _Float16* w1h = (_Float16*)carve(64 * 32 * 2);
    _Float16* w2h = (_Float16*)carve(128 * 128 * 2);
    _Float16* w3h = (_Float16*)carve(256 * 256 * 2);
    _Float16* woh = (_Float16*)carve(512 * 448 * 2);
    _Float16* xt0 = (_Float16*)carve((size_t)NPT * 32 * 2);
    _Float16* ht  = (_Float16*)carve((size_t)NPT * HTLD * 2);
    float*    xx  = (float*)carve((size_t)NPT * 4);
    int*      idx = (int*)carve((size_t)NPT * KNN * 4);
    float*    dist= (float*)carve((size_t)NPT * NPT * 4);
    _Float16* F   = (_Float16*)carve((size_t)NROWS * 256 * 2);
    float*    y   = (float*)carve((size_t)NROWS * 256 * 4);

    float* out = (float*)d_out;
    k_zero<<<(out_size + 255) / 256, 256, 0, stream>>>(out, out_size);

    k_cvt_w<<<(64 * 32 + 255) / 256, 256, 0, stream>>>(w1, w1h, 64, 6, 32);
    k_cvt_w<<<(128 * 128 + 255) / 256, 256, 0, stream>>>(w2, w2h, 128, 128, 128);
    k_cvt_w<<<(256 * 256 + 255) / 256, 256, 0, stream>>>(w3, w3h, 256, 256, 256);
    k_cvt_w<<<(512 * 448 + 255) / 256, 256, 0, stream>>>(wo, woh, 512, 448, 448);

    for (int b = 0; b < 8; ++b) {
        const float* xb = x + (size_t)b * 3 * NPT;
        // ----- layer 1 (C=3, feature 6->32 pad, O=64) -----
        k_pack_x0<<<NPT / 256, 256, 0, stream>>>(xb, xt0, xx);
        k_topk_l1<<<NPT, 32, 0, stream>>>(xb, xx, idx);
        k_feat<3, 32><<<(NROWS * 4 + 255) / 256, 256, 0, stream>>>(xt0, 32, idx, F);
        k_edge_gemm<32, 64><<<NROWS / 128, 256, 0, stream>>>(F, w1h, y);
        k_bn_max<<<(NPT * 64 + 255) / 256, 256, 0, stream>>>(y, 64, b1, g1, be1, m1, v1, ht + 0);
        // ----- layer 2 (C=64, feature 128, O=128) -----
        k_sumsq<<<NPT / 256, 256, 0, stream>>>(ht + 0, 64, xx);
        k_dist_gemm<64><<<dim3(16, 16), 256, 0, stream>>>(ht + 0, HTLD, xx, dist);
        k_topk<<<NPT, 32, 0, stream>>>(dist, idx);
        k_feat<64, 128><<<(NROWS * 16 + 255) / 256, 256, 0, stream>>>(ht + 0, HTLD, idx, F);
        k_edge_gemm<128, 128><<<NROWS / 128, 256, 0, stream>>>(F, w2h, y);
        k_bn_max<<<(NPT * 128 + 255) / 256, 256, 0, stream>>>(y, 128, b2, g2, be2, m2, v2, ht + 64);
        // ----- layer 3 (C=128, feature 256, O=256) -----
        k_sumsq<<<NPT / 256, 256, 0, stream>>>(ht + 64, 128, xx);
        k_dist_gemm<128><<<dim3(16, 16), 256, 0, stream>>>(ht + 64, HTLD, xx, dist);
        k_topk<<<NPT, 32, 0, stream>>>(dist, idx);
        k_feat<128, 256><<<(NROWS * 32 + 255) / 256, 256, 0, stream>>>(ht + 64, HTLD, idx, F);
        k_edge_gemm<256, 256><<<NROWS / 128, 256, 0, stream>>>(F, w3h, y);
        k_bn_max<<<(NPT * 256 + 255) / 256, 256, 0, stream>>>(y, 256, b3, g3, be3, m3, v3, ht + 192);
        // ----- final projection (512 x 448) + global max over N -----
        k_final<<<dim3(4, 8), 256, 0, stream>>>(woh, ht, bo, go, beo, mo, vo, out + b * 512);
    }
}
